// CausalSelfAttention_38646115730036
// MI455X (gfx1250) — matrix operand order repaired and checked by reading, not compiled, not run on hardware
//
#include <hip/hip_runtime.h>
#include <hip/hip_bf16.h>

#define B_ 2
#define T_ 4096
#define C_ 768
#define NH_ 12
#define HS_ 64
#define M_ROWS (B_*T_)      // 8192
#define N_QKV  (3*C_)       // 2304

typedef __bf16 bf16_t;
typedef __bf16 v16bf __attribute__((ext_vector_type(16)));
typedef __bf16 v8bf  __attribute__((ext_vector_type(8)));
typedef float  v8f   __attribute__((ext_vector_type(8)));

static __device__ inline v8bf ldv8(const bf16_t* p) { return *(const v8bf*)p; }

static __device__ inline v16bf cat16(v8bf a, v8bf b) {
  v16bf r;
#pragma unroll
  for (int i = 0; i < 8; ++i) { r[i] = a[i]; r[i + 8] = b[i]; }
  return r;
}

static __device__ inline v8f wmma_bf16(v16bf a, v16bf b, v8f c) {
  return __builtin_amdgcn_wmma_f32_16x16x32_bf16(false, a, false, b, (short)0, c,
                                                 false, false);
}

// ---- CDNA5 async global->LDS (ASYNCcnt-tracked, no VGPR data movement) ----
// Generic shared-aperture pointers carry the LDS byte offset in addr[31:0].
static __device__ inline void async_ld_b128(const void* g, void* lds) {
  asm volatile("global_load_async_to_lds_b128 %0, %1, off"
               :
               : "v"((unsigned)(uintptr_t)lds), "v"((unsigned long long)(uintptr_t)g)
               : "memory");
}
static __device__ inline void wait_async0() {
  asm volatile("s_wait_asynccnt 0" ::: "memory");
}

// ---------------------------------------------------------------- converts
__global__ __launch_bounds__(256) void f32_to_bf16_kernel(const float* __restrict__ in,
                                                          bf16_t* __restrict__ out,
                                                          int n) {
  for (int i = blockIdx.x * blockDim.x + threadIdx.x; i < n;
       i += gridDim.x * blockDim.x)
    out[i] = (bf16_t)in[i];
}

// in: [R][Ccols] f32  ->  out: [Ccols][R] bf16  (weights pre-transposed once
// so GEMM B-tiles can stream via async DMA with zero in-loop transposition)
__global__ __launch_bounds__(256) void transpose_f32_bf16(const float* __restrict__ in,
                                                          bf16_t* __restrict__ out,
                                                          int R, int Ccols) {
  __shared__ float tile[32][33];
  const int r0 = blockIdx.y * 32, c0 = blockIdx.x * 32;
  const int tx = threadIdx.x, ty = threadIdx.y;  // blockDim = (32, 8)
#pragma unroll
  for (int i = 0; i < 32; i += 8)
    tile[ty + i][tx] = in[(size_t)(r0 + ty + i) * Ccols + c0 + tx];
  __syncthreads();
#pragma unroll
  for (int i = 0; i < 32; i += 8)
    out[(size_t)(c0 + ty + i) * R + r0 + tx] = (bf16_t)tile[tx][ty + i];
}

// ---------------------------------------------------------------- GEMM
// C[M,N] = A[M,K](bf16, row-major) * Bt[N,K](bf16, pre-transposed) + bias
// MODE 0: scatter bf16 into q (pre-scaled by 1/sqrt(HS)), k, vt layouts
// MODE 1: f32 result to outF[M,N]
// Block: 128x128 tile, 256 threads (8 waves), wave = 32x64 subtile.
// Double-buffered LDS; both tiles staged with async-to-LDS DMA.
template <int MODE>
__global__ __launch_bounds__(256) void gemm_wmma(
    const bf16_t* __restrict__ A, const bf16_t* __restrict__ Bt,
    const float* __restrict__ bias, float* __restrict__ outF,
    bf16_t* __restrict__ qp, bf16_t* __restrict__ kp, bf16_t* __restrict__ vtp,
    int M, int N, int K) {
  __shared__ bf16_t lA[2][128 * 32];   // [m][k]
  __shared__ bf16_t lBt[2][128 * 32];  // [n][k]

  const int tid  = threadIdx.x;
  const int lane = tid & 31, wv = tid >> 5;
  const int half = lane >> 4, hh = lane & 15;
  const int m0 = blockIdx.y * 128, n0 = blockIdx.x * 128;
  const int wr = (wv >> 1) * 32, wc = (wv & 1) * 64;

  v8f acc[2][4];
#pragma unroll
  for (int i = 0; i < 2; ++i)
#pragma unroll
    for (int j = 0; j < 4; ++j)
#pragma unroll
      for (int e = 0; e < 8; ++e) acc[i][j][e] = 0.f;

  const int srow = tid >> 1, scol = (tid & 1) << 4;  // 128 rows x 32 cols

  auto stageA = [&](int bufi, int k0) {
    const bf16_t* ag = A + (size_t)(m0 + srow) * K + k0 + scol;
    async_ld_b128(ag,     &lA[bufi][srow * 32 + scol]);
    async_ld_b128(ag + 8, &lA[bufi][srow * 32 + scol + 8]);
  };
  auto stageB = [&](int bufi, int k0) {
    const bf16_t* bg = Bt + (size_t)(n0 + srow) * K + k0 + scol;
    async_ld_b128(bg,     &lBt[bufi][srow * 32 + scol]);
    async_ld_b128(bg + 8, &lBt[bufi][srow * 32 + scol + 8]);
  };

  stageA(0, 0);
  stageB(0, 0);

  const int KT = K / 32;
  for (int kt = 0; kt < KT; ++kt) {
    const int buf = kt & 1;
    wait_async0();
    __syncthreads();
    if (kt + 1 < KT) {
      stageA(buf ^ 1, (kt + 1) * 32);
      stageB(buf ^ 1, (kt + 1) * 32);
    }

    v16bf fa[2], fb[4];
#pragma unroll
    for (int tm = 0; tm < 2; ++tm) {
      const bf16_t* p = &lA[buf][(wr + tm * 16 + hh) * 32];
      fa[tm] = cat16(ldv8(p + half * 8), ldv8(p + 16 + half * 8));
    }
#pragma unroll
    for (int tn = 0; tn < 4; ++tn) {
      const bf16_t* p = &lBt[buf][(wc + tn * 16 + hh) * 32 + half * 8];
      fb[tn] = cat16(ldv8(p), ldv8(p + 16));
    }
#pragma unroll
    for (int tm = 0; tm < 2; ++tm)
#pragma unroll
      for (int tn = 0; tn < 4; ++tn)
        acc[tm][tn] = wmma_bf16(fa[tm], fb[tn], acc[tm][tn]);
  }

  // epilogue: C layout => lane holds col (hh), rows = half*8 + v
#pragma unroll
  for (int tm = 0; tm < 2; ++tm) {
#pragma unroll
    for (int tn = 0; tn < 4; ++tn) {
      const int gc  = n0 + wc + tn * 16 + hh;
      const float bv = bias[gc];
      const int grb = m0 + wr + tm * 16 + half * 8;
      if (MODE == 1) {
#pragma unroll
        for (int v = 0; v < 8; ++v)
          outF[(size_t)(grb + v) * N + gc] = acc[tm][tn][v] + bv;
      } else {
        const int which = gc / C_;
        const int rc = gc % C_;
        const int hidx = rc / HS_, d = rc % HS_;
#pragma unroll
        for (int v = 0; v < 8; ++v) {
          const int row = grb + v;
          const int bb = row / T_, tt = row % T_;
          const float val = acc[tm][tn][v] + bv;
          if (which == 0)  // fold softmax scale 1/sqrt(HS) into Q
            qp[((size_t)(bb * NH_ + hidx) * T_ + tt) * HS_ + d] =
                (bf16_t)(val * 0.125f);
          else if (which == 1)
            kp[((size_t)(bb * NH_ + hidx) * T_ + tt) * HS_ + d] = (bf16_t)val;
          else
            vtp[((size_t)(bb * NH_ + hidx) * HS_ + d) * T_ + tt] = (bf16_t)val;
        }
      }
    }
  }
}

// ---------------------------------------------------------------- flash attention
// grid = (T/128, B*NH), block = 256 (8 waves), wave owns 16 q rows.
// Double-buffered async K/Vt staging; mask only on diagonal tiles.
__global__ __launch_bounds__(256) void flash_attn(const bf16_t* __restrict__ q,
                                                  const bf16_t* __restrict__ k,
                                                  const bf16_t* __restrict__ vt,
                                                  bf16_t* __restrict__ y) {
  __shared__ bf16_t lK[2][64 * 64];    // [kv][d]
  __shared__ bf16_t lVt[2][64 * 64];   // [d][kv]
  __shared__ bf16_t lP[8 * 16 * 64];   // per-wave P staging

  const int tid  = threadIdx.x;
  const int lane = tid & 31, wv = tid >> 5;
  const int half = lane >> 4, hh = lane & 15;
  const int qt = blockIdx.x;
  const int bh = blockIdx.y;
  const int b = bh / NH_, h = bh % NH_;
  const int q0 = qt * 128 + wv * 16;  // wave's first q row (global within T)

  // Q fragments held for the whole kernel (A layout: lane row = hh)
  v16bf fq[2];
  {
    const bf16_t* qr = q + ((size_t)bh * T_ + q0 + hh) * HS_;
    fq[0] = cat16(ldv8(qr + half * 8),      ldv8(qr + 16 + half * 8));
    fq[1] = cat16(ldv8(qr + 32 + half * 8), ldv8(qr + 48 + half * 8));
  }

  v8f o[4], m_st, l_st;
#pragma unroll
  for (int i = 0; i < 8; ++i) { m_st[i] = -1e30f; l_st[i] = 0.f; }
#pragma unroll
  for (int dt = 0; dt < 4; ++dt)
#pragma unroll
    for (int i = 0; i < 8; ++i) o[dt][i] = 0.f;

  const int ldr = tid >> 2, ldc = (tid & 3) << 4;
  const int nkv = 2 * qt + 2;  // causal: kv tiles with base <= last q row
  bf16_t* lPw = lP + wv * (16 * 64);

  auto stage = [&](int bufi, int kv0) {
    const bf16_t* kg = k + ((size_t)bh * T_ + kv0 + ldr) * HS_ + ldc;
    async_ld_b128(kg,     &lK[bufi][ldr * 64 + ldc]);
    async_ld_b128(kg + 8, &lK[bufi][ldr * 64 + ldc + 8]);
    const bf16_t* vg = vt + ((size_t)bh * HS_ + ldr) * T_ + kv0 + ldc;
    async_ld_b128(vg,     &lVt[bufi][ldr * 64 + ldc]);
    async_ld_b128(vg + 8, &lVt[bufi][ldr * 64 + ldc + 8]);
  };

  stage(0, 0);

  for (int it = 0; it < nkv; ++it) {
    const int buf = it & 1;
    const int kv0 = it * 64;
    wait_async0();
    __syncthreads();
    if (it + 1 < nkv) stage(buf ^ 1, kv0 + 64);

    // wave-uniform skip: tile entirely above the diagonal contributes nothing
    if (kv0 <= q0 + 15) {
      // S = Q @ K^T  (B column n = K row => contiguous loads from lK)
      v8f s[4];
#pragma unroll
      for (int tn = 0; tn < 4; ++tn)
#pragma unroll
        for (int e = 0; e < 8; ++e) s[tn][e] = 0.f;
#pragma unroll
      for (int ks = 0; ks < 2; ++ks) {
#pragma unroll
        for (int tn = 0; tn < 4; ++tn) {
          const bf16_t* p = &lK[buf][(tn * 16 + hh) * 64 + ks * 32 + half * 8];
          v16bf fbv = cat16(ldv8(p), ldv8(p + 16));
          s[tn] = wmma_bf16(fq[ks], fbv, s[tn]);
        }
      }

      // causal mask only on diagonal tiles (scale already folded into Q)
      if (kv0 + 63 > q0) {
#pragma unroll
        for (int tn = 0; tn < 4; ++tn) {
          const int kc = kv0 + tn * 16 + hh;
#pragma unroll
          for (int v = 0; v < 8; ++v) {
            const int qr = q0 + half * 8 + v;
            s[tn][v] = (kc > qr) ? -1e30f : s[tn][v];
          }
        }
      }

      // row max over 64 kv cols
      v8f tmax = s[0];
#pragma unroll
      for (int tn = 1; tn < 4; ++tn)
#pragma unroll
        for (int e = 0; e < 8; ++e) tmax[e] = fmaxf(tmax[e], s[tn][e]);
      for (int off = 1; off < 16; off <<= 1) {
#pragma unroll
        for (int e = 0; e < 8; ++e)
          tmax[e] = fmaxf(tmax[e], __shfl_xor(tmax[e], off, 32));
      }

      v8f mnew, alpha;
#pragma unroll
      for (int e = 0; e < 8; ++e) {
        mnew[e]  = fmaxf(m_st[e], tmax[e]);
        alpha[e] = __expf(m_st[e] - mnew[e]);
      }

      // P = exp(S - mnew), row sum
      v8f rsum;
#pragma unroll
      for (int e = 0; e < 8; ++e) rsum[e] = 0.f;
#pragma unroll
      for (int tn = 0; tn < 4; ++tn)
#pragma unroll
        for (int v = 0; v < 8; ++v) {
          const float e = __expf(s[tn][v] - mnew[v]);
          s[tn][v] = e;
          rsum[v] += e;
        }
      for (int off = 1; off < 16; off <<= 1) {
#pragma unroll
        for (int e = 0; e < 8; ++e) rsum[e] += __shfl_xor(rsum[e], off, 32);
      }

#pragma unroll
      for (int e = 0; e < 8; ++e) l_st[e] = l_st[e] * alpha[e] + rsum[e];
#pragma unroll
      for (int dt = 0; dt < 4; ++dt)
#pragma unroll
        for (int e = 0; e < 8; ++e) o[dt][e] *= alpha[e];
      m_st = mnew;

      // re-layout P (C layout -> A layout) through per-wave LDS patch
#pragma unroll
      for (int tn = 0; tn < 4; ++tn)
#pragma unroll
        for (int v = 0; v < 8; ++v)
          lPw[(half * 8 + v) * 64 + tn * 16 + hh] = (bf16_t)s[tn][v];

      // O += P @ V  (Vt in LDS -> contiguous B-fragment loads)
#pragma unroll
      for (int ks = 0; ks < 2; ++ks) {
        const bf16_t* pp = lPw + hh * 64 + ks * 32;
        v16bf fav = cat16(ldv8(pp + half * 8), ldv8(pp + 16 + half * 8));
#pragma unroll
        for (int dt = 0; dt < 4; ++dt) {
          const bf16_t* vp = &lVt[buf][(dt * 16 + hh) * 64 + ks * 32 + half * 8];
          v16bf fbv = cat16(ldv8(vp), ldv8(vp + 16));
          o[dt] = wmma_bf16(fav, fbv, o[dt]);
        }
      }
    }
  }

  // normalize and store y[b, t, h*64 + d] (bf16 for the projection GEMM)
#pragma unroll
  for (int dt = 0; dt < 4; ++dt) {
#pragma unroll
    for (int v = 0; v < 8; ++v) {
      const int t = q0 + half * 8 + v;
      y[(size_t)(b * T_ + t) * C_ + h * HS_ + dt * 16 + hh] =
          (bf16_t)(o[dt][v] / l_st[v]);
    }
  }
}

// ---------------------------------------------------------------- launch
extern "C" void kernel_launch(void* const* d_in, const int* in_sizes, int n_in,
                              void* d_out, int out_size, void* d_ws, size_t ws_size,
                              hipStream_t stream) {
  (void)in_sizes; (void)n_in; (void)out_size; (void)ws_size;

  const float* x      = (const float*)d_in[0];
  const float* w_attn = (const float*)d_in[1];
  const float* b_attn = (const float*)d_in[2];
  const float* w_proj = (const float*)d_in[3];
  const float* b_proj = (const float*)d_in[4];
  float* out = (float*)d_out;

  char* ws = (char*)d_ws;
  size_t off = 0;
  auto alloc = [&](size_t bytes) -> void* {
    void* p = ws + off;
    off += (bytes + 255) & ~(size_t)255;
    return p;
  };
  bf16_t* xb   = (bf16_t*)alloc((size_t)M_ROWS * C_ * 2);
  bf16_t* wabt = (bf16_t*)alloc((size_t)C_ * N_QKV * 2);  // [N_QKV][C] transposed
  bf16_t* wpbt = (bf16_t*)alloc((size_t)C_ * C_ * 2);     // [C][C] transposed
  bf16_t* qb   = (bf16_t*)alloc((size_t)B_ * NH_ * T_ * HS_ * 2);
  bf16_t* kb   = (bf16_t*)alloc((size_t)B_ * NH_ * T_ * HS_ * 2);
  bf16_t* vtb  = (bf16_t*)alloc((size_t)B_ * NH_ * T_ * HS_ * 2);
  bf16_t* yb   = (bf16_t*)alloc((size_t)M_ROWS * C_ * 2);

  f32_to_bf16_kernel<<<1024, 256, 0, stream>>>(x, xb, M_ROWS * C_);
  transpose_f32_bf16<<<dim3(N_QKV / 32, C_ / 32), dim3(32, 8), 0, stream>>>(
      w_attn, wabt, C_, N_QKV);
  transpose_f32_bf16<<<dim3(C_ / 32, C_ / 32), dim3(32, 8), 0, stream>>>(
      w_proj, wpbt, C_, C_);

  gemm_wmma<0><<<dim3(N_QKV / 128, M_ROWS / 128), 256, 0, stream>>>(
      xb, wabt, b_attn, nullptr, qb, kb, vtb, M_ROWS, N_QKV, C_);

  flash_attn<<<dim3(T_ / 128, B_ * NH_), 256, 0, stream>>>(qb, kb, vtb, yb);

  gemm_wmma<1><<<dim3(C_ / 128, M_ROWS / 128), 256, 0, stream>>>(
      yb, wpbt, b_proj, out, nullptr, nullptr, nullptr, M_ROWS, C_, C_);
}
